// GNUMAP2_62852551410001
// MI455X (gfx1250) — matrix operand order repaired and checked by reading, not compiled, not run on hardware
//
#include <hip/hip_runtime.h>
#include <hip/hip_bf16.h>

typedef __attribute__((ext_vector_type(16))) _Float16 v16h;
typedef __attribute__((ext_vector_type(8)))  _Float16 v8h;
typedef __attribute__((ext_vector_type(8)))  float    v8f;
typedef __attribute__((ext_vector_type(4)))  unsigned int v4u;
typedef __attribute__((ext_vector_type(8)))  int      v8i;
typedef __attribute__((ext_vector_type(4)))  int      v4i;

#define N_NODES 8192
#define D_IN    512
#define D_HID   256
#define N_EDGES 262144
#define ALPHA_C 0.1520f
#define BETA_C  0.7900f

// ---------------- degree / dinv_sqrt ----------------
__global__ void k_init_deg(float* __restrict__ deg) {
    int i = blockIdx.x * blockDim.x + threadIdx.x;
    if (i < N_NODES) deg[i] = 1.0f;  // self loop
}

__global__ void k_count_deg(const int* __restrict__ dst, float* __restrict__ deg) {
    int e = blockIdx.x * blockDim.x + threadIdx.x;
    if (e < N_EDGES) atomicAdd(&deg[dst[e]], 1.0f);
}

__global__ void k_dinv(float* __restrict__ deg) {
    int i = blockIdx.x * blockDim.x + threadIdx.x;
    if (i < N_NODES) deg[i] = rsqrtf(deg[i]);
}

// ---------------- self-loop init: agg = x * dinv^2 ----------------
template <int D>
__global__ void k_self_init(const float* __restrict__ x, const float* __restrict__ dinv,
                            float* __restrict__ agg) {
    long idx = (long)blockIdx.x * blockDim.x + threadIdx.x;
    if (idx < (long)N_NODES * D) {
        int i = (int)(idx / D);
        float w = dinv[i];
        agg[idx] = x[idx] * (w * w);
    }
}

// ---------------- edge scatter: agg[dst] += x[src] * dinv[src]*dinv[dst] ----------------
template <int D>
__global__ void k_scatter(const int* __restrict__ src, const int* __restrict__ dst,
                          const float* __restrict__ x, const float* __restrict__ dinv,
                          float* __restrict__ agg) {
    int e = blockIdx.x;
    int s = src[e];
    int d = dst[e];
    float norm = dinv[s] * dinv[d];
    const float4* xs = (const float4*)(x + (long)s * D);
    float*        ad = agg + (long)d * D;
    int t = threadIdx.x;              // 0 .. D/4-1
    float4 v = xs[t];
    atomicAdd(&ad[4 * t + 0], v.x * norm);
    atomicAdd(&ad[4 * t + 1], v.y * norm);
    atomicAdd(&ad[4 * t + 2], v.z * norm);
    atomicAdd(&ad[4 * t + 3], v.w * norm);
}

// ---------------- f32 -> f16 conversions (one-time, feeds WMMA) ----------------
__global__ void k_cvt_A(const float* __restrict__ a32, _Float16* __restrict__ a16) {
    long idx = (long)blockIdx.x * blockDim.x + threadIdx.x;  // N_NODES*D_IN
    a16[idx] = (_Float16)a32[idx];
}

// W1[512k][256n] f32 -> W1T[256n][512k] f16 (so B fragments are contiguous)
__global__ void k_cvt_W1T(const float* __restrict__ w, _Float16* __restrict__ wt) {
    int idx = blockIdx.x * blockDim.x + threadIdx.x;   // 256*512
    int n = idx >> 9;          // /512
    int k = idx & 511;
    wt[idx] = (_Float16)w[k * D_HID + n];
}

// ---------------- GEMM1: H = relu(A16[8192,512] @ W1 + b), f16 WMMA ----------------
// 128 threads = 4 waves; block computes 64 rows x 16 cols.
// B tile (16 cols x full K=512, f16, 16KB) is DMA'd into LDS by the TDM once per block.
__global__ __launch_bounds__(128)
void k_gemm1_wmma(const _Float16* __restrict__ A16, const _Float16* __restrict__ W1T,
                  const float* __restrict__ bias, float* __restrict__ H) {
    __shared__ _Float16 shB[16 * D_IN];   // shB[n][k], n = 0..15 of this block's column tile

    const int tid  = threadIdx.x;
    const int wave = tid >> 5;
    const int lane = tid & 31;
    const int half = lane >> 4;       // 0 or 1
    const int l16  = lane & 15;
    const int m_base = blockIdx.x * 64 + wave * 16;
    const int n_base = blockIdx.y * 16;

    if (tid < 32) {
        // --- Tensor Data Mover: 2D tile (tile_dim0=512 K-elems, tile_dim1=16 rows) ---
        unsigned lds_off = (unsigned)(unsigned long long)(&shB[0]);  // addr[31:0] = LDS offset
        unsigned long long ga =
            (unsigned long long)(const void*)(W1T + (size_t)n_base * D_IN);
        v4u g0;
        g0[0] = 1u;                                    // count=1, is_restore=0, gather=0
        g0[1] = lds_off;                               // lds_addr
        g0[2] = (unsigned)ga;                          // global_addr[31:0]
        g0[3] = (unsigned)((ga >> 32) & 0x1FFFFFFull)  // global_addr[56:32]
                | (2u << 30);                          // type = 2 ("image")
        v8i g1;
        g1[0] = 0x10000;            // workgroup_mask=0, data_size=1 (2 bytes)
        g1[1] = (int)(512u << 16);  // tensor_dim0 = 512 (bits 79:48, low half)
        g1[2] = (int)(16u << 16);   // tensor_dim0 hi = 0, tensor_dim1 = 16 (bits 111:80, low half)
        g1[3] = (int)(512u << 16);  // tensor_dim1 hi = 0, tile_dim0 = 512 (bits 127:112)
        g1[4] = 16;                 // tile_dim1 = 16, tile_dim2 = 0
        g1[5] = 512;                // tensor_dim0_stride = 512 (bits 207:160, low bits)
        g1[6] = 0;
        g1[7] = 0;                  // tensor_dim1_stride = 0 (unused, 2D tile)
        v4i gz = {0, 0, 0, 0};
#if __clang_major__ >= 23
        v8i gz8 = {0, 0, 0, 0, 0, 0, 0, 0};
        __builtin_amdgcn_tensor_load_to_lds(g0, g1, gz, gz, gz8, 0);
#else
        __builtin_amdgcn_tensor_load_to_lds(g0, g1, gz, gz, 0);
#endif
        __builtin_amdgcn_s_wait_tensorcnt(0);
    }
    __syncthreads();

    v8f c = {};
    const _Float16* arow = A16 + (size_t)(m_base + l16) * D_IN;
    for (int k0 = 0; k0 < D_IN; k0 += 32) {
        __builtin_prefetch(arow + k0 + 128, 0, 1);   // global_prefetch_b8
        // A fragment (16-bit A 16x32 layout): VGPR 0-3 -> K = k0+8*half .. +7
        //                                     VGPR 4-7 -> K = k0+16+8*half .. +7
        v8h alo = *(const v8h*)(arow + k0 + 8 * half);
        v8h ahi = *(const v8h*)(arow + k0 + 16 + 8 * half);
        v16h a = __builtin_shufflevector(alo, ahi, 0, 1, 2, 3, 4, 5, 6, 7,
                                         8, 9, 10, 11, 12, 13, 14, 15);
        // B fragment (16-bit B 32x16 layout): lane n=l16 needs K = k0+8*half .. +7
        // and K = k0+16+8*half .. +7 in shB[n][k] -> two LDS b128 loads
        const v8h* bp = (const v8h*)(&shB[l16 * D_IN + k0 + 8 * half]);
        v8h blo = bp[0];
        v8h bhi = bp[2];
        v16h b = __builtin_shufflevector(blo, bhi, 0, 1, 2, 3, 4, 5, 6, 7,
                                         8, 9, 10, 11, 12, 13, 14, 15);
        c = __builtin_amdgcn_wmma_f32_16x16x32_f16(
                /*neg_a=*/false, a, /*neg_b=*/false, b,
                /*c_mod=*/(short)0, c, /*reuse_a=*/false, /*reuse_b=*/false);
    }

    // C/D layout: VGPR r -> lanes 0-15: M=r ; lanes 16-31: M=r+8 ; N = lane&15
    int n = n_base + l16;
    float bv = bias[n];
#pragma unroll
    for (int r = 0; r < 8; ++r) {
        int m = m_base + r + (half << 3);
        float val = c[r] + bv;
        H[(long)m * D_HID + n] = val > 0.0f ? val : 0.0f;
    }
}

// ---------------- GEMM2: emb = A2[8192,256] @ W2[256,2] + b2 ----------------
__global__ void k_gemm2(const float* __restrict__ A2, const float* __restrict__ W2,
                        const float* __restrict__ b2, float* __restrict__ emb) {
    int i = blockIdx.x * blockDim.x + threadIdx.x;
    if (i >= N_NODES) return;
    const float4* row = (const float4*)(A2 + (long)i * D_HID);
    float acc0 = b2[0], acc1 = b2[1];
#pragma unroll 2
    for (int k4 = 0; k4 < D_HID / 4; ++k4) {
        float4 v = row[k4];
        const float* w = W2 + 8 * k4;
        acc0 += v.x * w[0] + v.y * w[2] + v.z * w[4] + v.w * w[6];
        acc1 += v.x * w[1] + v.y * w[3] + v.z * w[5] + v.w * w[7];
    }
    emb[2 * i]     = acc0;
    emb[2 * i + 1] = acc1;
}

// ---------------- pairwise q, 4 outputs/thread, b128 stores ----------------
__global__ void k_pairwise(const float* __restrict__ emb, float4* __restrict__ q4) {
    long gid = (long)blockIdx.x * blockDim.x + threadIdx.x;   // over N*N/4
    long base = gid * 4;                                      // i*N + j0
    int i  = (int)(base >> 13);
    int j0 = (int)(base & (N_NODES - 1));
    float2 ei = ((const float2*)emb)[i];
    float4 out;
    float* o = &out.x;
#pragma unroll
    for (int t = 0; t < 4; ++t) {
        float2 ej = ((const float2*)emb)[j0 + t];
        float dx = ei.x - ej.x;
        float dy = ei.y - ej.y;
        float d2 = dx * dx + dy * dy;
        float qv = 1.0f;
        if (d2 > 0.0f) qv = 1.0f / (1.0f + ALPHA_C * __powf(d2, BETA_C));
        o[t] = qv;
    }
    q4[gid] = out;
}

extern "C" void kernel_launch(void* const* d_in, const int* in_sizes, int n_in,
                              void* d_out, int out_size, void* d_ws, size_t ws_size,
                              hipStream_t stream) {
    const float* features = (const float*)d_in[0];   // [8192,512]
    const int*   edge     = (const int*)d_in[1];     // [2,262144]
    const float* W1       = (const float*)d_in[2];   // [512,256]
    const float* b1       = (const float*)d_in[3];   // [256]
    const float* W2       = (const float*)d_in[4];   // [256,2]
    const float* b2       = (const float*)d_in[5];   // [2]
    const int* src = edge;
    const int* dst = edge + N_EDGES;

    float* out = (float*)d_out;
    float* emb = out;                      // 8192*2
    float* q   = out + 2 * N_NODES;        // 8192*8192

    float* ws   = (float*)d_ws;
    float* dinv = ws;                                  // 8192 f32
    float* agg1 = ws + N_NODES;                        // 8192*512 f32
    float* h    = agg1 + (long)N_NODES * D_IN;         // 8192*256 f32
    float* agg2 = h + (long)N_NODES * D_HID;           // 8192*256 f32
    _Float16* A16  = (_Float16*)(agg2 + (long)N_NODES * D_HID);   // 8192*512 f16
    _Float16* W1T  = A16 + (long)N_NODES * D_IN;                  // 256*512 f16

    // 1) degree + dinv_sqrt
    k_init_deg<<<N_NODES / 256, 256, 0, stream>>>(dinv);
    k_count_deg<<<N_EDGES / 256, 256, 0, stream>>>(dst, dinv);
    k_dinv<<<N_NODES / 256, 256, 0, stream>>>(dinv);

    // 2) propagate features (D=512)
    k_self_init<D_IN><<<(N_NODES * D_IN) / 256, 256, 0, stream>>>(features, dinv, agg1);
    k_scatter<D_IN><<<N_EDGES, D_IN / 4, 0, stream>>>(src, dst, features, dinv, agg1);

    // 3) convert to f16 (A row-major, W1 transposed) then layer-1 WMMA GEMM + relu
    k_cvt_A<<<(N_NODES * D_IN) / 256, 256, 0, stream>>>(agg1, A16);
    k_cvt_W1T<<<(D_HID * D_IN) / 256, 256, 0, stream>>>(W1, W1T);
    dim3 g1(N_NODES / 64, D_HID / 16);
    k_gemm1_wmma<<<g1, 128, 0, stream>>>(A16, W1T, b1, h);

    // 4) propagate hidden (D=256)
    k_self_init<D_HID><<<(N_NODES * D_HID) / 256, 256, 0, stream>>>(h, dinv, agg2);
    k_scatter<D_HID><<<N_EDGES, D_HID / 4, 0, stream>>>(src, dst, h, dinv, agg2);

    // 5) layer 2 GEMM -> emb (written straight to d_out)
    k_gemm2<<<N_NODES / 256, 256, 0, stream>>>(agg2, W2, b2, emb);

    // 6) pairwise q (dominant cost: 268 MB store stream, b128 stores)
    long total4 = (long)N_NODES * N_NODES / 4;
    k_pairwise<<<(unsigned)(total4 / 256), 256, 0, stream>>>(emb, (float4*)q);
}
